// M2_MOEP_28312424415468
// MI455X (gfx1250) — matrix operand order repaired and checked by reading, not compiled, not run on hardware
//
#include <hip/hip_runtime.h>
#include <stdint.h>

// ---------------------------------------------------------------------------
// CDNA5 (gfx1250) types / helpers
// ---------------------------------------------------------------------------
typedef __attribute__((ext_vector_type(16))) __bf16        v16bf;
typedef __attribute__((ext_vector_type(8)))  __bf16        v8bf;
typedef __attribute__((ext_vector_type(8)))  float         v8f;
typedef __attribute__((ext_vector_type(8)))  unsigned int  u32x8;
typedef __attribute__((ext_vector_type(4)))  unsigned int  u32x4;

__device__ __forceinline__ __bf16 f32_to_bf16(float f) {
    unsigned int u = __builtin_bit_cast(unsigned int, f);
    unsigned int rnd = 0x7FFFu + ((u >> 16) & 1u);   // round-to-nearest-even
    u += rnd;
    unsigned short h = (unsigned short)(u >> 16);
    return __builtin_bit_cast(__bf16, h);
}

#define BF16_WMMA(a, b, c) \
    __builtin_amdgcn_wmma_f32_16x16x32_bf16(false, (a), false, (b), (short)0, (c), false, false)
#define CAT16(lo, hi) \
    __builtin_shufflevector((lo), (hi), 0,1,2,3,4,5,6,7,8,9,10,11,12,13,14,15)

// ---------------------------------------------------------------------------
// Elementwise f32 -> bf16 pack (row-major), RNE.
// ---------------------------------------------------------------------------
__global__ __launch_bounds__(256) void pack_bf16(const float* __restrict__ in,
                                                 __bf16* __restrict__ out, long n)
{
    long i = (long)blockIdx.x * 256 + threadIdx.x;
    if (i < n) out[i] = f32_to_bf16(in[i]);
}

// ---------------------------------------------------------------------------
// Pack a (K x N) f32 weight into WMMA-fragment-major bf16:
//   out[((kt*tilesN + nt)*32 + lane)*16 + i] = B[(kt*32 + (lane>>4)*8 + i + (i&8))*ldb
//                                               + nt*16 + (lane&15)]
// so a GEMM wave reads its whole 32x16 B fragment as ONE 32-byte vector load.
// gridDim.y = batch (expert) index.
// ---------------------------------------------------------------------------
__global__ __launch_bounds__(256) void pack_bfrag(
    const float* __restrict__ B, long ldb, long in_bstride,
    __bf16* __restrict__ out, long out_bstride, int K, int N)
{
    int  tilesN = N >> 4;
    long t      = (long)blockIdx.x * 256 + threadIdx.x;
    long total  = (long)(K >> 5) * tilesN * 32;
    if (t >= total) return;
    int  lane = (int)(t & 31);
    long blk  = t >> 5;
    int  nt   = (int)(blk % tilesN);
    int  kt   = (int)(blk / tilesN);
    int  half = lane >> 4, ncol = lane & 15;
    const float* src = B + (long)blockIdx.y * in_bstride
                         + ((long)kt * 32 + half * 8) * ldb + nt * 16 + ncol;
    v16bf vals;
#pragma unroll
    for (int i = 0; i < 16; i++) vals[i] = f32_to_bf16(src[(long)(i + (i & 8)) * ldb]);
    *(v16bf*)(out + (long)blockIdx.y * out_bstride + t * 16) = vals;
}

// ---------------------------------------------------------------------------
// Generic WMMA GEMM: C[M,N] = A[M,K](bf16 row-major) @ Bp(fragment-packed bf16)
//                    + bias, f32 out. One wave = one 16x16 C tile.
// A fragment = two contiguous 8xbf16 runs (2x b128 loads); B = one 32B load.
// ---------------------------------------------------------------------------
__global__ __launch_bounds__(128) void wmma_gemm_pk(
    const __bf16* __restrict__ A, long lda,
    const __bf16* __restrict__ Bp,
    const float* __restrict__ bias,
    float* __restrict__ C, long ldc,
    int M, int N, int K)
{
    int tilesN = N >> 4;
    int tiles  = (M >> 4) * tilesN;
    int waveId = blockIdx.x * (blockDim.x >> 5) + (threadIdx.x >> 5);
    if (waveId >= tiles) return;
    int tm = waveId / tilesN, tn = waveId % tilesN;
    int lane = threadIdx.x & 31, half = lane >> 4, mrow = lane & 15, ncol = lane & 15;

    const __bf16* arow    = A + ((long)tm * 16 + mrow) * lda;
    const __bf16* bbase   = Bp + ((long)tn * 32 + lane) * 16;
    long          bstride = (long)tilesN * 512;     // elements per k-tile step

    v8f acc = {};
    int kts = K >> 5;
    for (int kt = 0; kt < kts; kt++) {
        v8bf alo = *(const v8bf*)(arow + kt * 32 + half * 8);
        v8bf ahi = *(const v8bf*)(arow + kt * 32 + 16 + half * 8);
        v16bf a  = CAT16(alo, ahi);
        v16bf b  = *(const v16bf*)(bbase + (long)kt * bstride);
        acc = BF16_WMMA(a, b, acc);
    }
    float bv = bias ? bias[tn * 16 + ncol] : 0.0f;
#pragma unroll
    for (int r = 0; r < 8; r++)
        C[((long)tm * 16 + r + half * 8) * ldc + tn * 16 + ncol] = acc[r] + bv;
}

// ---------------------------------------------------------------------------
// Gating fc1: h1(32,512) = xbf(32,32768) @ gW1(f32 32768x512) + gb1.
// Memory-bound on the one-shot 64MB gW1 read -> convert f32->bf16 in-register
// (one v_perm_b32 per element pair), reuse each B fragment for both M tiles.
// 32 waves total (one per N tile).
// ---------------------------------------------------------------------------
__global__ __launch_bounds__(128) void wmma_gemm_fc1(
    const __bf16* __restrict__ A,      // (32, 32768) bf16
    const float* __restrict__ B,       // (32768, 512) f32
    const float* __restrict__ bias,
    float* __restrict__ C)             // (32, 512) f32
{
    int tn = blockIdx.x * (blockDim.x >> 5) + (threadIdx.x >> 5);
    if (tn >= 32) return;
    int lane = threadIdx.x & 31, half = lane >> 4, mrow = lane & 15, ncol = lane & 15;

    const __bf16* a0   = A + (long)mrow * 32768;
    const __bf16* a1   = A + (long)(16 + mrow) * 32768;
    const float*  bcol = B + (long)(half * 16) * 512 + tn * 16 + ncol;

    v8f acc0 = {}, acc1 = {};
    for (int k0 = 0; k0 < 32768; k0 += 32) {
        __builtin_prefetch(bcol + (long)(k0 + 64) * 512, 0, 1);
        u32x8 bu;
#pragma unroll
        for (int i = 0; i < 8; i++) {
            unsigned f0 = __builtin_bit_cast(unsigned, bcol[(long)(k0 + 2 * i) * 512]);
            unsigned f1 = __builtin_bit_cast(unsigned, bcol[(long)(k0 + 2 * i + 1) * 512]);
            bu[i] = __builtin_amdgcn_perm(f1, f0, 0x07060302u);   // {bf16(f1),bf16(f0)}
        }
        v16bf b = __builtin_bit_cast(v16bf, bu);

        v8bf lo0 = *(const v8bf*)(a0 + k0 + half * 16);
        v8bf hi0 = *(const v8bf*)(a0 + k0 + half * 16 + 8);
        acc0 = BF16_WMMA(CAT16(lo0, hi0), b, acc0);
        v8bf lo1 = *(const v8bf*)(a1 + k0 + half * 16);
        v8bf hi1 = *(const v8bf*)(a1 + k0 + half * 16 + 8);
        acc1 = BF16_WMMA(CAT16(lo1, hi1), b, acc1);
    }
    float bv = bias[tn * 16 + ncol];
#pragma unroll
    for (int r = 0; r < 8; r++) {
        C[(long)(r + half * 8) * 512 + tn * 16 + ncol]      = acc0[r] + bv;
        C[(long)(16 + r + half * 8) * 512 + tn * 16 + ncol] = acc1[r] + bv;
    }
}

// ---------------------------------------------------------------------------
// Row LayerNorm (+ optional ReLU); optional f32 and/or bf16 outputs.
// One 256-thread block per row. In-place safe (yf may alias X).
// ---------------------------------------------------------------------------
__global__ __launch_bounds__(256) void layernorm_dual(
    const float* __restrict__ X, const float* __restrict__ g,
    const float* __restrict__ bta, float* yf, __bf16* yb,
    int n, int do_relu)
{
    long row = blockIdx.x;
    const float* x = X + row * n;
    __shared__ float red[256], red2[256], mv[2];
    int tid = threadIdx.x;
    float s = 0.f, s2 = 0.f;
    for (int i = tid; i < n; i += 256) { float v = x[i]; s += v; s2 += v * v; }
    red[tid] = s; red2[tid] = s2;
    __syncthreads();
    for (int off = 128; off > 0; off >>= 1) {
        if (tid < off) { red[tid] += red[tid + off]; red2[tid] += red2[tid + off]; }
        __syncthreads();
    }
    if (tid == 0) {
        float m = red[0] / n, vr = red2[0] / n - (red[0] / n) * (red[0] / n);
        mv[0] = m; mv[1] = rsqrtf(vr + 1e-5f);
    }
    __syncthreads();
    float m = mv[0], rs = mv[1];
    for (int i = tid; i < n; i += 256) {
        float v = (x[i] - m) * rs * g[i] + bta[i];
        if (do_relu) v = v > 0.f ? v : 0.f;
        if (yf) yf[row * n + i] = v;
        if (yb) yb[row * n + i] = f32_to_bf16(v);
    }
}

// ---------------------------------------------------------------------------
// Gate weights: prototype distances + softmax. emb:(32,128) protos:(8,128)
// ---------------------------------------------------------------------------
__global__ void gate_weights_kernel(const float* __restrict__ emb,
                                    const float* __restrict__ protos,
                                    const float* __restrict__ temp,
                                    float* __restrict__ w)
{
    int b = threadIdx.x;
    if (b >= 32) return;
    float t = fminf(fmaxf(temp[0], 0.1f), 5.0f);
    float logits[8], mx = -1e30f;
    for (int e = 0; e < 8; e++) {
        float d2 = 0.f;
        for (int k = 0; k < 128; k++) {
            float d = emb[b * 128 + k] - protos[e * 128 + k];
            d2 += d * d;
        }
        logits[e] = -sqrtf(fmaxf(d2, 1e-12f)) / t;
        mx = fmaxf(mx, logits[e]);
    }
    float sum = 0.f;
    for (int e = 0; e < 8; e++) { logits[e] = __expf(logits[e] - mx); sum += logits[e]; }
    float inv = 1.0f / sum;
    for (int e = 0; e < 8; e++) w[b * 8 + e] = logits[e] * inv;
}

// ---------------------------------------------------------------------------
// Fused expert stage, LAST TIMESTEP ONLY (s=511):
//   fusedb[b,:] = sum_e (w[b,e]*relu(x[b,511,:]@eW1[e]+eb1)) @ eW2[e]
//               + sum_e  w[b,e]*eb2[e]
// Single workgroup, 8 waves. M=32 (2 m-tiles), N=512 (4 n-tiles/wave).
// x-tile (32x64 bf16, 4KB) staged by the Tensor Data Mover; he (32x512 bf16,
// 32KB) lives in LDS; fused accumulators persist in VGPRs across experts.
// ---------------------------------------------------------------------------
__global__ __launch_bounds__(256) void expert_fused_last(
    const __bf16* __restrict__ xbf,    // (32, 32768) bf16
    const __bf16* __restrict__ eW1p,   // packed, per e: 64x512
    const float* __restrict__ eb1,     // (8,512)
    const __bf16* __restrict__ eW2p,   // packed, per e: 512x512
    const float* __restrict__ eb2,     // (8,512)
    const float* __restrict__ wg,      // (32,8)
    __bf16* __restrict__ fusedb)       // (32,512) bf16
{
    __shared__ __bf16 x_lds[32][64];    //  4 KB
    __shared__ __bf16 he_lds[32][512];  // 32 KB
    __shared__ float  wg_lds[256];

    int tid = threadIdx.x, wave = tid >> 5, lane = tid & 31;
    int half = lane >> 4, mrow = lane & 15, ncol = lane & 15;

    wg_lds[tid] = wg[tid];              // 32x8 gate weights

    if (wave == 0) {
        // Tensor Data Mover: 2D tile, 2B elems, 64 cols x 32 rows,
        // row stride 32768 elems, from xbf[:, 511*64 ..] into x_lds.
        unsigned lds_off = (unsigned)(unsigned long long)(void*)&x_lds[0][0];
        unsigned long long ga = (unsigned long long)(const void*)(xbf + 511 * 64);
        u32x4 g0;
        g0[0] = 1u;                                  // count=1, user descriptor
        g0[1] = lds_off;                             // lds_addr
        g0[2] = (unsigned)ga;                        // global_addr[31:0]
        g0[3] = ((unsigned)(ga >> 32) & 0x1FFFFFFu) | (2u << 30);  // addr hi | type=2
        u32x8 g1;
        g1[0] = 0x00010000u;     // workgroup_mask=0, data_size=1 (2B)
        g1[1] = 64u << 16;       // tensor_dim0 = 64
        g1[2] = 32u << 16;       // tensor_dim1 = 32
        g1[3] = 64u << 16;       // tile_dim0 = 64
        g1[4] = 32u;             // tile_dim1 = 32, tile_dim2 = 0
        g1[5] = 32768u;          // tensor_dim0_stride = 32768 elems
        g1[6] = 0u;
        g1[7] = 0u;
        asm volatile("tensor_load_to_lds %0, %1" :: "s"(g0), "s"(g1) : "memory");
        __builtin_amdgcn_s_wait_tensorcnt(0);
    }
    __syncthreads();

    v8f facc[2][4] = {};

    for (int e = 0; e < 8; e++) {
        const __bf16* W1p = eW1p + (long)e * (64 * 512);
        const __bf16* W2p = eW2p + (long)e * (512 * 512);

        // he = w[b,e] * relu(x @ W1 + b1) -> LDS (bf16)
#pragma unroll
        for (int mt = 0; mt < 2; mt++) {
#pragma unroll
            for (int j = 0; j < 4; j++) {
                int nt = wave * 4 + j, n0 = nt << 4;
                v8f hacc = {};
#pragma unroll
                for (int kc = 0; kc < 2; kc++) {
                    const __bf16* ar = &x_lds[mt * 16 + mrow][kc * 32];
                    v8bf alo = *(const v8bf*)(ar + half * 8);
                    v8bf ahi = *(const v8bf*)(ar + 16 + half * 8);
                    v16bf b  = *(const v16bf*)(W1p + ((long)(kc * 32 + nt) * 32 + lane) * 16);
                    hacc = BF16_WMMA(CAT16(alo, ahi), b, hacc);
                }
                float bv = eb1[e * 512 + n0 + ncol];
#pragma unroll
                for (int r = 0; r < 8; r++) {
                    int row = mt * 16 + r + half * 8;
                    float v = hacc[r] + bv;
                    v = v > 0.f ? v : 0.f;
                    he_lds[row][n0 + ncol] = f32_to_bf16(v * wg_lds[row * 8 + e]);
                }
            }
        }
        __syncthreads();

        // fused += he (32x512) @ W2 (512x512)
#pragma unroll
        for (int mt = 0; mt < 2; mt++) {
#pragma unroll
            for (int j = 0; j < 4; j++) {
                int nt = wave * 4 + j, n0 = nt << 4;
                v8f acc = facc[mt][j];
                for (int kc = 0; kc < 16; kc++) {
                    const __bf16* ar = &he_lds[mt * 16 + mrow][kc * 32];
                    v8bf alo = *(const v8bf*)(ar + half * 8);
                    v8bf ahi = *(const v8bf*)(ar + 16 + half * 8);
                    v16bf b  = *(const v16bf*)(W2p + ((long)(kc * 32 + nt) * 32 + lane) * 16);
                    acc = BF16_WMMA(CAT16(alo, ahi), b, acc);
                }
                float b2v = eb2[e * 512 + n0 + ncol];
#pragma unroll
                for (int r = 0; r < 8; r++)
                    acc[r] += b2v * wg_lds[(mt * 16 + r + half * 8) * 8 + e];
                facc[mt][j] = acc;
            }
        }
        __syncthreads();
    }

#pragma unroll
    for (int mt = 0; mt < 2; mt++)
#pragma unroll
        for (int j = 0; j < 4; j++) {
            int n0 = (wave * 4 + j) << 4;
#pragma unroll
            for (int r = 0; r < 8; r++)
                fusedb[(long)(mt * 16 + r + half * 8) * 512 + n0 + ncol] =
                    f32_to_bf16(facc[mt][j][r]);
        }
}

// ---------------------------------------------------------------------------
// Host orchestration
// ---------------------------------------------------------------------------
extern "C" void kernel_launch(void* const* d_in, const int* in_sizes, int n_in,
                              void* d_out, int out_size, void* d_ws, size_t ws_size,
                              hipStream_t stream)
{
    (void)in_sizes; (void)n_in; (void)out_size; (void)ws_size;

    const float* x      = (const float*)d_in[0];
    const float* gW1    = (const float*)d_in[1];
    const float* gb1    = (const float*)d_in[2];
    const float* gln1_g = (const float*)d_in[3];
    const float* gln1_b = (const float*)d_in[4];
    const float* gW2    = (const float*)d_in[5];
    const float* gb2    = (const float*)d_in[6];
    const float* gln2_g = (const float*)d_in[7];
    const float* gln2_b = (const float*)d_in[8];
    const float* gW3    = (const float*)d_in[9];
    const float* gb3    = (const float*)d_in[10];
    const float* protos = (const float*)d_in[11];
    const float* temp   = (const float*)d_in[12];
    const float* eW1    = (const float*)d_in[13];
    const float* eb1    = (const float*)d_in[14];
    const float* eW2    = (const float*)d_in[15];
    const float* eb2    = (const float*)d_in[16];
    const float* fW1    = (const float*)d_in[17];
    const float* fb1    = (const float*)d_in[18];
    const float* fln_g  = (const float*)d_in[19];
    const float* fln_b  = (const float*)d_in[20];
    const float* fW2    = (const float*)d_in[21];
    const float* fb2    = (const float*)d_in[22];
    const float* ln_g   = (const float*)d_in[23];
    const float* ln_b   = (const float*)d_in[24];
    const float* pW1    = (const float*)d_in[25];
    const float* pb1    = (const float*)d_in[26];
    const float* pln_g  = (const float*)d_in[27];
    const float* pln_b  = (const float*)d_in[28];
    const float* pW2    = (const float*)d_in[29];
    const float* pb2    = (const float*)d_in[30];
    const float* on_g   = (const float*)d_in[31];
    const float* on_b   = (const float*)d_in[32];

    uint8_t* wp = (uint8_t*)d_ws;
    auto take = [&](size_t bytes) -> uint8_t* {
        uint8_t* r = wp; wp += (bytes + 255) & ~(size_t)255; return r;
    };
    __bf16* xbf   = (__bf16*)take((size_t)32 * 32768 * 2);   // 2 MB
    __bf16* gW2p  = (__bf16*)take((size_t)512 * 512 * 2);
    __bf16* gW3p  = (__bf16*)take((size_t)512 * 128 * 2);
    __bf16* eW1p  = (__bf16*)take((size_t)8 * 64 * 512 * 2);
    __bf16* eW2p  = (__bf16*)take((size_t)8 * 512 * 512 * 2);
    __bf16* fW1p  = (__bf16*)take((size_t)512 * 512 * 2);
    __bf16* fW2p  = (__bf16*)take((size_t)512 * 512 * 2);
    __bf16* pW1p  = (__bf16*)take((size_t)512 * 256 * 2);
    __bf16* pW2p  = (__bf16*)take((size_t)256 * 6144 * 2);
    float*  h1    = (float*)take(32 * 512 * 4);
    __bf16* h1b   = (__bf16*)take(32 * 512 * 2);
    float*  h2    = (float*)take(32 * 512 * 4);
    __bf16* h2b   = (__bf16*)take(32 * 512 * 2);
    float*  emb   = (float*)take(32 * 128 * 4);
    float*  wsm   = (float*)take(32 * 8 * 4);
    __bf16* fusedb= (__bf16*)take(32 * 512 * 2);
    float*  t1    = (float*)take(32 * 512 * 4);
    __bf16* t1b   = (__bf16*)take(32 * 512 * 2);
    float*  f2    = (float*)take(32 * 512 * 4);
    __bf16* fb    = (__bf16*)take(32 * 512 * 2);
    float*  p1    = (float*)take(32 * 256 * 4);
    __bf16* p1b   = (__bf16*)take(32 * 256 * 2);
    float*  outp  = (float*)d_out;

    // ---- one-time per-call packs (bf16 / fragment-major) ---------------
    pack_bf16<<<4096, 256, 0, stream>>>(x, xbf, (long)32 * 32768);
    pack_bfrag<<<dim3(64, 1), 256, 0, stream>>>(gW2, 512, 0, gW2p, 0, 512, 512);
    pack_bfrag<<<dim3(16, 1), 256, 0, stream>>>(gW3, 128, 0, gW3p, 0, 512, 128);
    pack_bfrag<<<dim3(8, 8),  256, 0, stream>>>(eW1, 512, 64 * 512, eW1p, 64 * 512, 64, 512);
    pack_bfrag<<<dim3(64, 8), 256, 0, stream>>>(eW2, 512, 512 * 512, eW2p, 512 * 512, 512, 512);
    pack_bfrag<<<dim3(64, 1), 256, 0, stream>>>(fW1, 512, 0, fW1p, 0, 512, 512);
    pack_bfrag<<<dim3(64, 1), 256, 0, stream>>>(fW2, 512, 0, fW2p, 0, 512, 512);
    pack_bfrag<<<dim3(32, 1), 256, 0, stream>>>(pW1, 256, 0, pW1p, 0, 512, 256);
    pack_bfrag<<<dim3(384, 1),256, 0, stream>>>(pW2, 6144, 0, pW2p, 0, 256, 6144);

    // ---- gating network (full sequence needed) -------------------------
    wmma_gemm_fc1<<<8, 128, 0, stream>>>(xbf, gW1, gb1, h1);
    layernorm_dual<<<32, 256, 0, stream>>>(h1, gln1_g, gln1_b, nullptr, h1b, 512, 1);
    wmma_gemm_pk<<<16, 128, 0, stream>>>(h1b, 512, gW2p, gb2, h2, 512, 32, 512, 512);
    layernorm_dual<<<32, 256, 0, stream>>>(h2, gln2_g, gln2_b, nullptr, h2b, 512, 1);
    wmma_gemm_pk<<<4, 128, 0, stream>>>(h2b, 512, gW3p, gb3, emb, 128, 32, 128, 512);
    gate_weights_kernel<<<1, 32, 0, stream>>>(emb, protos, temp, wsm);

    // ---- expert + fusion path: only s = 511 reaches the output ---------
    expert_fused_last<<<1, 256, 0, stream>>>(xbf, eW1p, eb1, eW2p, eb2, wsm, fusedb);

    wmma_gemm_pk<<<16, 128, 0, stream>>>(fusedb, 512, fW1p, fb1, t1, 512, 32, 512, 512);
    layernorm_dual<<<32, 256, 0, stream>>>(t1, fln_g, fln_b, nullptr, t1b, 512, 1);
    wmma_gemm_pk<<<16, 128, 0, stream>>>(t1b, 512, fW2p, fb2, f2, 512, 32, 512, 512);
    layernorm_dual<<<32, 256, 0, stream>>>(f2, ln_g, ln_b, nullptr, fb, 512, 0);

    // ---- prediction head ----------------------------------------------
    wmma_gemm_pk<<<8, 128, 0, stream>>>(fb, 512, pW1p, pb1, p1, 256, 32, 256, 512);
    layernorm_dual<<<32, 256, 0, stream>>>(p1, pln_g, pln_b, nullptr, p1b, 256, 1);
    wmma_gemm_pk<<<192, 128, 0, stream>>>(p1b, 256, pW2p, pb2, outp, 6144, 32, 6144, 256);
    layernorm_dual<<<32 * 96, 256, 0, stream>>>(outp, on_g, on_b, outp, nullptr, 64, 0);
}
